// BaseVQ_11897059410176
// MI455X (gfx1250) — matrix operand order repaired and checked by reading, not compiled, not run on hardware
//
#include <hip/hip_runtime.h>
#include <hip/hip_bf16.h>
#include <stdint.h>

typedef __attribute__((ext_vector_type(16))) _Float16 v16h;
typedef __attribute__((ext_vector_type(8)))  _Float16 v8h;
typedef __attribute__((ext_vector_type(8)))  float    v8f;
typedef __attribute__((ext_vector_type(4)))  unsigned int v4u;
typedef __attribute__((ext_vector_type(8)))  int      v8i;
typedef __attribute__((ext_vector_type(4)))  int      v4i;

#define NROWS 65536
#define VOCAB 2048
#define EMBED 512
#define ZDIM  64

// LDS row pitch for staged emb chunk: 512 f16 (1024B) + 16B TDM pad -> bank spread
#define ROWPITCH 1040
#define BUF_BYTES (16 * ROWPITCH)   // 16640 B per chunk buffer

__device__ __forceinline__ v16h cat8(v8h lo, v8h hi) {
    return __builtin_shufflevector(lo, hi, 0,1,2,3,4,5,6,7,8,9,10,11,12,13,14,15);
}

__device__ __forceinline__ v8f wmma_f16(v16h a, v16h b, v8f c) {
    return __builtin_amdgcn_wmma_f32_16x16x32_f16(false, a, false, b, (short)0, c, false, false);
}

// ---------------------------------------------------------------------------
// TDM: stage one 16-row x 512-f16 codebook chunk into LDS (2D tile, padded).
// D# per CDNA5 ISA. 6-arg builtin on this
// toolchain: (v4u g0, v8i g1, v4i g2, v4i g3, v8i extra, i32 cpol).
// ---------------------------------------------------------------------------
__device__ __forceinline__ void tdm_load_chunk(const _Float16* emb_h, int cc,
                                               uint32_t lds_byte_addr) {
    uint64_t ga = (uint64_t)(uintptr_t)(emb_h + (size_t)cc * 16 * EMBED);
    v4u g0;
    g0[0] = 1u;                                   // count=1 (valid), user mode
    g0[1] = lds_byte_addr;                        // lds_addr
    g0[2] = (uint32_t)ga;                         // global_addr[31:0]
    g0[3] = (uint32_t)((ga >> 32) & 0x01FFFFFFu)  // global_addr[56:32]
          | 0x80000000u;                          // type = 2 ("image")
    v8i g1;
    g1[0] = (1 << 16)       // data_size = 2 bytes
          | (1 << 20)       // pad_enable
          | (7 << 22)       // pad_interval: every 256 dwords (one 512-f16 row)
          | (3 << 25);      // pad_amount: 4 dwords (16B) -> pitch 1040B
    g1[1] = (int)(512u << 16);     // tensor_dim0 = 512 (bits 79:48)
    g1[2] = (int)(2048u << 16);    // tensor_dim1 = 2048 (bits 111:80)
    g1[3] = (int)(512u << 16);     // tile_dim0 = 512 (bits 127:112)
    g1[4] = 16;                    // tile_dim1 = 16 rows; tile_dim2 = 0
    g1[5] = 512;                   // tensor_dim0_stride = 512 elements
    g1[6] = 0;                     // tensor_dim1_stride lo16 (ignored: tile_dim2=0)
    g1[7] = 16;                    // tensor_dim1_stride hi = (512*2048)>>16
    v4i gz4 = {0, 0, 0, 0};        // groups 2/3 unused (<=2D tensor)
    v8i gz8 = {0, 0, 0, 0, 0, 0, 0, 0};
    __builtin_amdgcn_tensor_load_to_lds(g0, g1, gz4, gz4, gz8, 0);
}

// ---------------------------------------------------------------------------
// Kernel 0a: f32 -> f16 conversions; pre/post weights stored transposed so the
// WMMA B operand (lane = column, 16 contiguous k per lane) is a contiguous load.
// ---------------------------------------------------------------------------
__global__ void k_convert(const float* __restrict__ emb_W,
                          const float* __restrict__ pre_W,
                          const float* __restrict__ post_W,
                          _Float16* __restrict__ emb_h,
                          _Float16* __restrict__ pre_Wt,
                          _Float16* __restrict__ post_Wt) {
    int i = blockIdx.x * blockDim.x + threadIdx.x;
    int stride = gridDim.x * blockDim.x;
    for (int t = i; t < VOCAB * EMBED; t += stride)
        emb_h[t] = (_Float16)emb_W[t];
    for (int t = i; t < ZDIM * EMBED; t += stride) {
        int k = t / EMBED, e = t - k * EMBED;        // pre_W[k][e]
        pre_Wt[e * ZDIM + k] = (_Float16)pre_W[t];
    }
    for (int t = i; t < EMBED * ZDIM; t += stride) {
        int k = t / ZDIM, j = t - k * ZDIM;          // post_W[k][j]
        post_Wt[j * EMBED + k] = (_Float16)post_W[t];
    }
}

// Kernel 0b: half squared norms of codebook rows (wave per row, shfl reduce).
__global__ void k_embnorm(const float* __restrict__ emb_W,
                          float* __restrict__ emb_hn) {
    int wave = (blockIdx.x * blockDim.x + threadIdx.x) >> 5;
    int lane = threadIdx.x & 31;
    if (wave >= VOCAB) return;
    const float* row = emb_W + (size_t)wave * EMBED;
    float s = 0.f;
    for (int k = lane; k < EMBED; k += 32) { float x = row[k]; s += x * x; }
    for (int m = 16; m >= 1; m >>= 1) s += __shfl_xor(s, m, 32);
    if (lane == 0) emb_hn[wave] = 0.5f * s;
}

// ---------------------------------------------------------------------------
// Kernel 1: zp = z @ pre_W + pre_b, emitted as f16 [N,512]. One wave = 16 rows.
// ---------------------------------------------------------------------------
__global__ void __launch_bounds__(256) k_pre(const float* __restrict__ z,
                                             const _Float16* __restrict__ pre_Wt,
                                             const float* __restrict__ pre_b,
                                             _Float16* __restrict__ zp_h) {
    const int lane = threadIdx.x & 31;
    const int wave = threadIdx.x >> 5;
    const int rowbase = (blockIdx.x * 8 + wave) * 16;
    const int r  = rowbase + (lane & 15);
    const int k8 = (lane >> 4) * 16;  // A operand k-offset (contiguous, matches B)
    // A: 16x64 f32 tile -> two 16x32 f16 WMMA operands, register resident
    v16h a[2];
#pragma unroll
    for (int kc = 0; kc < 2; ++kc) {
        const float* p0 = z + (size_t)r * ZDIM + kc * 32 + k8;
        v8h lo, hi;
#pragma unroll
        for (int j = 0; j < 8; ++j) {
            lo[j] = (_Float16)p0[j];
            hi[j] = (_Float16)p0[8 + j];
        }
        a[kc] = cat8(lo, hi);
    }
    const int k16 = (lane >> 4) * 16; // B operand k-offset
    const int mo  = (lane >> 4) * 8;  // D operand row offset
    for (int ct = 0; ct < EMBED / 16; ++ct) {
        const int col = ct * 16 + (lane & 15);
        v8f acc = {};
#pragma unroll
        for (int kc = 0; kc < 2; ++kc) {
            const _Float16* bp = pre_Wt + (size_t)col * ZDIM + kc * 32 + k16;
            v16h b = cat8(*(const v8h*)bp, *(const v8h*)(bp + 8));
            acc = wmma_f16(a[kc], b, acc);
        }
        float bias = pre_b[col];
#pragma unroll
        for (int v = 0; v < 8; ++v)
            zp_h[(size_t)(rowbase + mo + v) * EMBED + col] = (_Float16)(acc[v] + bias);
    }
}

// ---------------------------------------------------------------------------
// Kernel 2: nearest-code search. argmin |zp-e|^2 == argmax (zp.e - 0.5|e|^2).
// Wave = 16 rows (A resident, 128 VGPRs). Block = 8 waves shares codebook
// chunks staged in LDS by the TDM, double buffered.
// ---------------------------------------------------------------------------
__global__ void __launch_bounds__(256) k_vq(const _Float16* __restrict__ zp_h,
                                            const _Float16* __restrict__ emb_h,
                                            const float* __restrict__ emb_hn,
                                            int* __restrict__ tokens,
                                            float* __restrict__ tok_out) {
    __shared__ char smem[2 * BUF_BYTES];
    const int lane = threadIdx.x & 31;
    const int wave = threadIdx.x >> 5;
    const int rowbase = (blockIdx.x * 8 + wave) * 16;
    const int r  = rowbase + (lane & 15);
    const int k8 = (lane >> 4) * 16;
    const uint32_t lds_base = (uint32_t)(uintptr_t)(&smem[0]);

    if (threadIdx.x == 0)               // prologue: stage chunk 0 (TENSORcnt)
        tdm_load_chunk(emb_h, 0, lds_base);

    // A: 16 rows x 512 f16, fully register resident (16 x v16h)
    v16h a[16];
#pragma unroll
    for (int kc = 0; kc < 16; ++kc) {
        const _Float16* p = zp_h + (size_t)r * EMBED + kc * 32 + k8;
        a[kc] = cat8(*(const v8h*)p, *(const v8h*)(p + 8));
    }

    float best[8];
    int   bidx[8];
#pragma unroll
    for (int v = 0; v < 8; ++v) { best[v] = -3.4e38f; bidx[v] = 0; }

    const int k16 = (lane >> 4) * 16;
    for (int cc = 0; cc < VOCAB / 16; ++cc) {
        __builtin_amdgcn_s_wait_tensorcnt(0);   // chunk cc landed (wave0; nop elsewhere)
        __syncthreads();                        // publish LDS + retire prev buffer use
        if (cc + 1 < VOCAB / 16 && threadIdx.x == 0)
            tdm_load_chunk(emb_h, cc + 1, lds_base + ((cc + 1) & 1) * BUF_BYTES);

        const char* buf = smem + (cc & 1) * BUF_BYTES + (lane & 15) * ROWPITCH;
        v8f acc = {};
#pragma unroll
        for (int kc = 0; kc < 16; ++kc) {
            const _Float16* bp = (const _Float16*)(buf + (kc * 32 + k16) * 2);
            v16h b = cat8(*(const v8h*)bp, *(const v8h*)(bp + 8));
            acc = wmma_f16(a[kc], b, acc);
        }
        const int   n  = cc * 16 + (lane & 15);
        const float hn = emb_hn[n];
#pragma unroll
        for (int v = 0; v < 8; ++v) {
            float sc = acc[v] - hn;             // zp.e - 0.5|e|^2
            if (sc > best[v]) { best[v] = sc; bidx[v] = n; }
        }
    }

    // Cross-lane argmax within each 16-lane half (D layout: n = lane&15).
#pragma unroll
    for (int m = 1; m < 16; m <<= 1) {
#pragma unroll
        for (int v = 0; v < 8; ++v) {
            float os = __shfl_xor(best[v], m, 32);
            int   oi = __shfl_xor(bidx[v], m, 32);
            if (os > best[v] || (os == best[v] && oi < bidx[v])) {
                best[v] = os; bidx[v] = oi;
            }
        }
    }
    if ((lane & 15) == 0) {
        const int mo = (lane >> 4) * 8;
#pragma unroll
        for (int v = 0; v < 8; ++v) {
            tokens[rowbase + mo + v]  = bidx[v];
            tok_out[rowbase + mo + v] = (float)bidx[v];
        }
    }
}

// ---------------------------------------------------------------------------
// Kernel 3: z_q = emb[tok] @ post_W + post_b. Wave = 16 rows, K=512, 64 cols.
// ---------------------------------------------------------------------------
__global__ void __launch_bounds__(256) k_post(const int* __restrict__ tokens,
                                              const _Float16* __restrict__ emb_h,
                                              const _Float16* __restrict__ post_Wt,
                                              const float* __restrict__ post_b,
                                              float* __restrict__ zq_out) {
    const int lane = threadIdx.x & 31;
    const int wave = threadIdx.x >> 5;
    const int rowbase = (blockIdx.x * 8 + wave) * 16;
    const int tok = tokens[rowbase + (lane & 15)];
    const int k8  = (lane >> 4) * 16;
    v16h a[16];                       // gathered codebook row, register resident
#pragma unroll
    for (int kc = 0; kc < 16; ++kc) {
        const _Float16* p = emb_h + (size_t)tok * EMBED + kc * 32 + k8;
        a[kc] = cat8(*(const v8h*)p, *(const v8h*)(p + 8));
    }
    const int k16 = (lane >> 4) * 16;
    const int mo  = (lane >> 4) * 8;
    for (int jt = 0; jt < ZDIM / 16; ++jt) {
        const int col = jt * 16 + (lane & 15);
        v8f acc = {};
#pragma unroll
        for (int kc = 0; kc < 16; ++kc) {
            const _Float16* bp = post_Wt + (size_t)col * EMBED + kc * 32 + k16;
            v16h b = cat8(*(const v8h*)bp, *(const v8h*)(bp + 8));
            acc = wmma_f16(a[kc], b, acc);
        }
        float bias = post_b[col];
#pragma unroll
        for (int v = 0; v < 8; ++v)
            zq_out[(size_t)(rowbase + mo + v) * ZDIM + col] = acc[v] + bias;
    }
}

// ---------------------------------------------------------------------------
extern "C" void kernel_launch(void* const* d_in, const int* in_sizes, int n_in,
                              void* d_out, int out_size, void* d_ws, size_t ws_size,
                              hipStream_t stream) {
    (void)in_sizes; (void)n_in; (void)out_size; (void)ws_size;
    const float* z      = (const float*)d_in[0];
    const float* emb_W  = (const float*)d_in[1];
    const float* pre_W  = (const float*)d_in[2];
    const float* pre_b  = (const float*)d_in[3];
    const float* post_W = (const float*)d_in[4];
    const float* post_b = (const float*)d_in[5];

    float* out     = (float*)d_out;
    float* tok_out = out;                // [N] tokens (as float, tuple part 0)
    float* zq_out  = out + NROWS;        // [N,64] z_q (tuple part 1)

    char* ws = (char*)d_ws;
    _Float16* emb_h   = (_Float16*)(ws);                                  // 2 MB
    _Float16* pre_Wt  = (_Float16*)(ws + (2u << 20));                     // 64 KB
    _Float16* post_Wt = (_Float16*)(ws + (2u << 20) + (64u << 10));       // 64 KB
    float*    emb_hn  = (float*)   (ws + (2u << 20) + (128u << 10));      // 8 KB
    int*      tokens  = (int*)     (ws + (2u << 20) + (136u << 10));      // 256 KB
    _Float16* zp_h    = (_Float16*)(ws + (4u << 20));                     // 64 MB

    k_convert<<<dim3(512), dim3(256), 0, stream>>>(emb_W, pre_W, post_W,
                                                   emb_h, pre_Wt, post_Wt);
    k_embnorm<<<dim3(VOCAB / 8), dim3(256), 0, stream>>>(emb_W, emb_hn);
    k_pre <<<dim3(NROWS / 128), dim3(256), 0, stream>>>(z, pre_Wt, pre_b, zp_h);
    k_vq  <<<dim3(NROWS / 128), dim3(256), 0, stream>>>(zp_h, emb_h, emb_hn,
                                                        tokens, tok_out);
    k_post<<<dim3(NROWS / 128), dim3(256), 0, stream>>>(tokens, emb_h, post_Wt,
                                                        post_b, zq_out);
}
